// MultiAttBlock_85341000171624
// MI455X (gfx1250) — matrix operand order repaired and checked by reading, not compiled, not run on hardware
//
#include <hip/hip_runtime.h>

// Problem constants (match reference)
#define B_ 4
#define L_ 512
#define S_ 4096
#define E_ 512
#define H_ 8
#define D_ 64
#define SCALE_ 0.125f   // D^-0.5

typedef __attribute__((ext_vector_type(16))) __bf16 bf16x16;
typedef __attribute__((ext_vector_type(8)))  __bf16 bf16x8;
typedef __attribute__((ext_vector_type(8)))  float  f32x8;
typedef __attribute__((ext_vector_type(4)))  float  f32x4;

// ---------------------------------------------------------------------------
// Weight prep: f32 [K,N] -> bf16 transposed [N,K] (tiny, 256K elems per mat)
// ---------------------------------------------------------------------------
__global__ __launch_bounds__(256)
void convert_wT(const float* __restrict__ W, __bf16* __restrict__ WT)
{
    int idx = blockIdx.x * 256 + threadIdx.x;      // linear over output [N,K]
    if (idx >= E_ * E_) return;
    int n = idx / E_;
    int k = idx % E_;
    WT[idx] = (__bf16)W[(long long)k * E_ + n];
}

// ---------------------------------------------------------------------------
// Generic BF16-WMMA GEMM: C[z] = act( A_f32[z][M,K] x BT_bf16[z][N,K]^T )
//   z = blockIdx.z, decomposed as (zb = z/hDiv, zh = z%hDiv) for per-(b,h)
//   addressing. A is converted f32->bf16 in registers during fragment load.
//   Options: bias[N], residual (C-layout f32), scale, key-padding mask
//   (mask[col]==0 -> -inf), output f32 / bf16 / bf16-transposed.
// Block: 256 threads = 8 waves; block tile 128(M) x 64(N); wave tile 32x32.
// All M,N multiples of 128/64 resp., K multiple of 32 (true for every call).
// ---------------------------------------------------------------------------
__global__ __launch_bounds__(256)
void wmma_gemm_kernel(
    const float*  __restrict__ A,  long long sbA, long long shA, int lda,
    const __bf16* __restrict__ BT, long long sbB, long long shB, int ldb,
    const float*  __restrict__ bias,
    const float*  __restrict__ resid, long long sbR, long long shR,
    const int*    __restrict__ mask,  long long sbM,
    float scale,
    float*  __restrict__ Cf, __bf16* __restrict__ Cb,
    long long sbC, long long shC, int ldc,
    int M, int N, int K, int hDiv, int transC)
{
    const int z  = blockIdx.z;
    const int zb = z / hDiv;
    const int zh = z % hDiv;

    const float*  Ab = A  + (long long)zb * sbA + (long long)zh * shA;
    const __bf16* Bb = BT + (long long)zb * sbB + (long long)zh * shB;
    const float*  Rb = resid ? resid + (long long)zb * sbR + (long long)zh * shR : nullptr;
    const int*    Mb = mask  ? mask  + (long long)zb * sbM : nullptr;
    const long long cOff = (long long)zb * sbC + (long long)zh * shC;

    const int tid  = threadIdx.x;
    const int lane = tid & 31;
    const int wave = tid >> 5;
    const int r    = lane & 15;   // row-in-tile for A / col-in-tile for B,C
    const int half = lane >> 4;

    const int m0 = blockIdx.y * 128 + (wave >> 1) * 32;
    const int n0 = blockIdx.x * 64  + (wave & 1)  * 32;

    f32x8 acc[2][2] = {};

    for (int k0 = 0; k0 < K; k0 += 32) {
        bf16x16 a[2], b[2];

        // A fragments (16x32 bf16, converted from f32):
        // per-lane K order: half0 -> {0..15}, half1 -> {16..31}
#pragma unroll
        for (int t = 0; t < 2; ++t) {
            const float* ap = Ab + (long long)(m0 + t * 16 + r) * lda + k0 + half * 16;
            f32x4 c0a = *reinterpret_cast<const f32x4*>(ap);
            f32x4 c0b = *reinterpret_cast<const f32x4*>(ap + 4);
            f32x4 c1a = *reinterpret_cast<const f32x4*>(ap + 8);
            f32x4 c1b = *reinterpret_cast<const f32x4*>(ap + 12);
            bf16x16 v;
#pragma unroll
            for (int i = 0; i < 4; ++i) {
                v[i]      = (__bf16)c0a[i];
                v[4 + i]  = (__bf16)c0b[i];
                v[8 + i]  = (__bf16)c1a[i];
                v[12 + i] = (__bf16)c1b[i];
            }
            a[t] = v;
        }

        // B fragments (32x16 bf16): lane = col N, lanes 0-15 hold K 0..15,
        // lanes 16-31 hold K 16..31 (contiguous 32B load from BT[N,K])
#pragma unroll
        for (int t = 0; t < 2; ++t) {
            const __bf16* bp = Bb + (long long)(n0 + t * 16 + r) * ldb + k0 + half * 16;
            b[t] = *reinterpret_cast<const bf16x16*>(bp);
        }

        // Unconditional speculative prefetch of next K-step (SCOPE_DEV
        // prefetches are speculative; OOB translations are silently dropped,
        // so no guard -> no branch inside the hot loop).
        __builtin_prefetch(Ab + (long long)(m0 + r) * lda + k0 + 32, 0, 1);
        __builtin_prefetch(Bb + (long long)(n0 + r) * ldb + k0 + 32, 0, 1);

        acc[0][0] = __builtin_amdgcn_wmma_f32_16x16x32_bf16(false, a[0], false, b[0], 0, acc[0][0], false, false);
        acc[0][1] = __builtin_amdgcn_wmma_f32_16x16x32_bf16(false, a[0], false, b[1], 0, acc[0][1], false, false);
        acc[1][0] = __builtin_amdgcn_wmma_f32_16x16x32_bf16(false, a[1], false, b[0], 0, acc[1][0], false, false);
        acc[1][1] = __builtin_amdgcn_wmma_f32_16x16x32_bf16(false, a[1], false, b[1], 0, acc[1][1], false, false);
    }

    // C fragment: per-lane fixed col (n0 + tj*16 + r), 8 rows (half*8 + i)
#pragma unroll
    for (int ti = 0; ti < 2; ++ti)
#pragma unroll
        for (int tj = 0; tj < 2; ++tj) {
            const int  cn   = n0 + tj * 16 + r;
            const int  rm   = m0 + ti * 16 + half * 8;
            const float bval = bias ? bias[cn] : 0.0f;
            const bool  moff = Mb && (Mb[cn] == 0);

            if (transC) {
                // bf16 store into C[n, m] (contiguous along m) — used for V^T
                bf16x8 ov;
#pragma unroll
                for (int i = 0; i < 8; ++i) {
                    float v = acc[ti][tj][i] * scale + bval;
                    ov[i] = (__bf16)v;
                }
                *reinterpret_cast<bf16x8*>(Cb + cOff + (long long)cn * ldc + rm) = ov;
            } else if (Cf) {
#pragma unroll
                for (int i = 0; i < 8; ++i) {
                    float v = acc[ti][tj][i] * scale + bval;
                    if (Rb)  v += Rb[(long long)(rm + i) * ldc + cn];
                    if (moff) v = -__builtin_inff();
                    Cf[cOff + (long long)(rm + i) * ldc + cn] = v;
                }
            } else {
#pragma unroll
                for (int i = 0; i < 8; ++i) {
                    float v = acc[ti][tj][i] * scale + bval;
                    Cb[cOff + (long long)(rm + i) * ldc + cn] = (__bf16)v;
                }
            }
        }
}

// ---------------------------------------------------------------------------
// In-place softmax over rows of length S. One 256-thread block per row,
// 16 elements per thread held in registers; LDS tree reductions.
// ---------------------------------------------------------------------------
__global__ __launch_bounds__(256)
void softmax_rows(float* __restrict__ attn)
{
    __shared__ float red[256];
    float* p = attn + (long long)blockIdx.x * S_;
    const int t = threadIdx.x;

    f32x4 v[4];
    float mx = -__builtin_inff();
#pragma unroll
    for (int i = 0; i < 4; ++i) {
        v[i] = *reinterpret_cast<const f32x4*>(p + i * 1024 + t * 4);
#pragma unroll
        for (int j = 0; j < 4; ++j) mx = fmaxf(mx, v[i][j]);
    }
    red[t] = mx;
    __syncthreads();
    for (int s = 128; s > 0; s >>= 1) {
        if (t < s) red[t] = fmaxf(red[t], red[t + s]);
        __syncthreads();
    }
    mx = red[0];
    __syncthreads();

    float sum = 0.0f;
#pragma unroll
    for (int i = 0; i < 4; ++i)
#pragma unroll
        for (int j = 0; j < 4; ++j) {
            float e = __expf(v[i][j] - mx);   // exp(-inf - mx) == 0 for masked
            v[i][j] = e;
            sum += e;
        }
    red[t] = sum;
    __syncthreads();
    for (int s = 128; s > 0; s >>= 1) {
        if (t < s) red[t] += red[t + s];
        __syncthreads();
    }
    const float inv = 1.0f / red[0];

#pragma unroll
    for (int i = 0; i < 4; ++i) {
#pragma unroll
        for (int j = 0; j < 4; ++j) v[i][j] *= inv;
        *reinterpret_cast<f32x4*>(p + i * 1024 + t * 4) = v[i];
    }
}

// ---------------------------------------------------------------------------
extern "C" void kernel_launch(void* const* d_in, const int* in_sizes, int n_in,
                              void* d_out, int out_size, void* d_ws, size_t ws_size,
                              hipStream_t stream)
{
    const float* Q   = (const float*)d_in[0];
    const float* K   = (const float*)d_in[1];
    const float* V   = (const float*)d_in[2];
    const int*   kpm = (const int*)  d_in[3];
    const float* Wq  = (const float*)d_in[4];
    const float* bq  = (const float*)d_in[5];
    const float* Wk  = (const float*)d_in[6];
    const float* bk  = (const float*)d_in[7];
    const float* Wv  = (const float*)d_in[8];
    const float* bv  = (const float*)d_in[9];
    const float* Wo  = (const float*)d_in[10];
    const float* Wqb = (const float*)d_in[11];

    float* out  = (float*)d_out;                       // [B,L,E]
    float* attn = out + (size_t)B_ * L_ * E_;          // [B,H,L,S]

    // workspace carve-up (~47 MB, all sizes 256B-aligned by construction)
    char*  ws = (char*)d_ws;
    size_t o  = 0;
    __bf16* WqT  = (__bf16*)(ws + o); o += (size_t)E_ * E_ * 2;
    __bf16* WkT  = (__bf16*)(ws + o); o += (size_t)E_ * E_ * 2;
    __bf16* WvT  = (__bf16*)(ws + o); o += (size_t)E_ * E_ * 2;
    __bf16* WoT  = (__bf16*)(ws + o); o += (size_t)E_ * E_ * 2;
    __bf16* WqbT = (__bf16*)(ws + o); o += (size_t)E_ * E_ * 2;
    float*  qp   = (float*)(ws + o);  o += (size_t)B_ * L_ * E_ * 4;   // q_proj f32
    __bf16* kb   = (__bf16*)(ws + o); o += (size_t)B_ * S_ * E_ * 2;   // k bf16 [b,s,hd]
    __bf16* vT   = (__bf16*)(ws + o); o += (size_t)B_ * E_ * S_ * 2;   // v bf16 [b,hd,s]
    float*  ctx  = (float*)(ws + o);  o += (size_t)B_ * L_ * E_ * 4;   // attn context
    float*  qres = (float*)(ws + o);  o += (size_t)B_ * L_ * E_ * 4;   // _Q

    const int cvtBlocks = (E_ * E_ + 255) / 256;
    convert_wT<<<cvtBlocks, 256, 0, stream>>>(Wq,  WqT);
    convert_wT<<<cvtBlocks, 256, 0, stream>>>(Wk,  WkT);
    convert_wT<<<cvtBlocks, 256, 0, stream>>>(Wv,  WvT);
    convert_wT<<<cvtBlocks, 256, 0, stream>>>(Wo,  WoT);
    convert_wT<<<cvtBlocks, 256, 0, stream>>>(Wqb, WqbT);

    // q_proj = Q @ Wq + bq  -> f32 [B*L, E]
    wmma_gemm_kernel<<<dim3(E_ / 64, (B_ * L_) / 128, 1), 256, 0, stream>>>(
        Q, 0, 0, E_,  WqT, 0, 0, E_,  bq,
        nullptr, 0, 0,  nullptr, 0,  1.0f,
        qp, nullptr, 0, 0, E_,  B_ * L_, E_, E_, 1, 0);

    // k = K @ Wk + bk  -> bf16 [B*S, E]
    wmma_gemm_kernel<<<dim3(E_ / 64, (B_ * S_) / 128, 1), 256, 0, stream>>>(
        K, 0, 0, E_,  WkT, 0, 0, E_,  bk,
        nullptr, 0, 0,  nullptr, 0,  1.0f,
        nullptr, kb, 0, 0, E_,  B_ * S_, E_, E_, 1, 0);

    // v = V @ Wv + bv  -> bf16 transposed [b][hd][s]  (contiguous store path)
    wmma_gemm_kernel<<<dim3(E_ / 64, S_ / 128, B_), 256, 0, stream>>>(
        V, (long long)S_ * E_, 0, E_,  WvT, 0, 0, E_,  bv,
        nullptr, 0, 0,  nullptr, 0,  1.0f,
        nullptr, vT, (long long)E_ * S_, 0, S_,  S_, E_, E_, 1, 1);

    // scores[b,h] = SCALE * q[b,:,h,:] @ k[b,:,h,:]^T, mask -> -inf
    wmma_gemm_kernel<<<dim3(S_ / 64, L_ / 128, B_ * H_), 256, 0, stream>>>(
        qp, (long long)L_ * E_, D_, E_,
        kb, (long long)S_ * E_, D_, E_,
        nullptr,  nullptr, 0, 0,
        kpm, S_,  SCALE_,
        attn, nullptr, (long long)H_ * L_ * S_, (long long)L_ * S_, S_,
        L_, S_, D_, H_, 0);

    // softmax in place over S (this normalized tensor IS the attn_weights output)
    softmax_rows<<<B_ * H_ * L_, 256, 0, stream>>>(attn);

    // ctx[b,l,h*D+d] = attn[b,h,l,:] @ v[b,:,h,d]   (K = S = 4096)
    wmma_gemm_kernel<<<dim3(1, L_ / 128, B_ * H_), 256, 0, stream>>>(
        attn, (long long)H_ * L_ * S_, (long long)L_ * S_, S_,
        vT,   (long long)H_ * D_ * S_, (long long)D_ * S_, S_,
        nullptr,  nullptr, 0, 0,  nullptr, 0,  1.0f,
        ctx, nullptr, (long long)L_ * E_, D_, E_,
        L_, D_, S_, H_, 0);

    // _Q = ctx @ Wo + q_proj
    wmma_gemm_kernel<<<dim3(E_ / 64, (B_ * L_) / 128, 1), 256, 0, stream>>>(
        ctx, 0, 0, E_,  WoT, 0, 0, E_,  nullptr,
        qp, 0, 0,  nullptr, 0,  1.0f,
        qres, nullptr, 0, 0, E_,  B_ * L_, E_, E_, 1, 0);

    // out = _Q + _Q @ Wqb
    wmma_gemm_kernel<<<dim3(E_ / 64, (B_ * L_) / 128, 1), 256, 0, stream>>>(
        qres, 0, 0, E_,  WqbT, 0, 0, E_,  nullptr,
        qres, 0, 0,  nullptr, 0,  1.0f,
        out, nullptr, 0, 0, E_,  B_ * L_, E_, E_, 1, 0);
}
